// EncoderLayer_72258529788143
// MI455X (gfx1250) — matrix operand order repaired and checked by reading, not compiled, not run on hardware
//
#include <hip/hip_runtime.h>

// ---------------------------------------------------------------------------
// CDNA5 (gfx1250) bf16-WMMA transformer encoder forward pass.
// Wave32; WMMA 16x16x32 bf16 (f32 accumulate); GEMM staging via
// GLOBAL_LOAD_ASYNC_TO_LDS_B128 double buffering when the toolchain has it.
// ---------------------------------------------------------------------------

typedef __attribute__((ext_vector_type(16))) __bf16 v16bf;
typedef __attribute__((ext_vector_type(8)))  float  v8f;
typedef __attribute__((ext_vector_type(4)))  unsigned int v4u;
typedef int v4i_vs __attribute__((vector_size(16)));   // matches builtin param type

union BFrag { v16bf v; v4u u[2]; };

constexpr int B_  = 4;
constexpr int S_  = 1024;
constexpr int D_  = 1024;
constexpr int H_  = 16;
constexpr int HS_ = 64;
constexpr int FF_ = 4096;
constexpr int L_  = 6;
constexpr int MTOT = B_ * S_;   // 4096 token rows

#if defined(__gfx1250__) && __has_builtin(__builtin_amdgcn_global_load_async_to_lds_b128)
#define USE_ASYNC_LDS 1
#else
#define USE_ASYNC_LDS 0
#endif

static __device__ __forceinline__ unsigned short f2bf(float f) {
  unsigned int u = __float_as_uint(f);
  u += 0x7fffu + ((u >> 16) & 1u);          // round-to-nearest-even
  return (unsigned short)(u >> 16);
}

static __device__ __forceinline__ v8f wmma_bf16(const BFrag& a, const BFrag& b, v8f c) {
  // (neg_a, A, neg_b, B, c_mod, C, reuse_a, reuse_b)
  return __builtin_amdgcn_wmma_f32_16x16x32_bf16(false, a.v, false, b.v,
                                                 (short)0, c, false, false);
}

#if USE_ASYNC_LDS
// 16-byte async global -> LDS copy (ASYNCcnt-tracked, bypasses VGPRs).
// Builtin signature (from hipcc diagnostic): (v4i AS1*, v4i AS3*, Ii off, Ii cpol).
static __device__ __forceinline__ void async_copy16(const unsigned short* g,
                                                    unsigned short* l) {
  __builtin_amdgcn_global_load_async_to_lds_b128(
      (__attribute__((address_space(1))) v4i_vs*)const_cast<unsigned short*>(g),
      (__attribute__((address_space(3))) v4i_vs*)l,
      0, 0);
}
template <int N>
static __device__ __forceinline__ void wait_async() {
#if __has_builtin(__builtin_amdgcn_s_wait_asynccnt)
  __builtin_amdgcn_s_wait_asynccnt(N);
#else
  asm volatile("s_wait_asynccnt %0" :: "i"(N) : "memory");
#endif
}
#endif

// ---------------------------------------------------------------------------
// Weight convert + transpose: W[K][N] f32 -> Wt[N][K] bf16 (contiguous K rows
// so WMMA B-operand fragments are two 16B loads per lane, same as A).
// ---------------------------------------------------------------------------
__global__ __launch_bounds__(256) void wcvt_kernel(const float* __restrict__ W,
                                                   unsigned short* __restrict__ Wt,
                                                   int K, int N) {
  const size_t idx = (size_t)blockIdx.x * 256 + threadIdx.x;
  if (idx >= (size_t)K * N) return;
  const int k = (int)(idx / N);
  const int n = (int)(idx % N);
  Wt[(size_t)n * K + k] = f2bf(W[idx]);
}

// ---------------------------------------------------------------------------
// Embedding + sinusoidal positional encoding -> x (f32 residual) + xb (bf16).
// ---------------------------------------------------------------------------
__global__ __launch_bounds__(256) void embed_kernel(const int* __restrict__ toks,
                                                    const float* __restrict__ emb,
                                                    float* __restrict__ x,
                                                    unsigned short* __restrict__ xb) {
  const int rs  = blockIdx.x;          // b*S + s
  const int s   = rs & (S_ - 1);
  const int tok = toks[rs];
#pragma unroll
  for (int i = 0; i < 4; ++i) {
    const int d = threadIdx.x + i * 256;
    const float rate = __powf(10000.0f, (float)(d & ~1) / (float)D_);
    const float ang  = (float)s / rate;
    const float pe   = (d & 1) ? __cosf(ang) : __sinf(ang);
    const float val  = emb[(size_t)tok * D_ + d] + pe;
    x [(size_t)rs * D_ + d] = val;
    xb[(size_t)rs * D_ + d] = f2bf(val);
  }
}

// ---------------------------------------------------------------------------
// Tiled bf16 WMMA GEMM: C[M][N] = A[M][K] * Bt[N][K]^T + bias.
// 256 threads = 8 waves in a 2x4 grid; 128x128 tile; K-step 32; double-
// buffered LDS staged by async global->LDS copies (one tile of prefetch).
// MODE: 0 = f32 row-major, 1 = bf16 row-major, 2 = relu->bf16 row-major,
//       3 = bf16 scattered to per-head V^T layout [B][H][HS][S].
// ---------------------------------------------------------------------------
template <int MODE>
__global__ __launch_bounds__(256) void gemm_bf16_kernel(
    const unsigned short* __restrict__ A, const unsigned short* __restrict__ Bt,
    const float* __restrict__ bias, void* __restrict__ Cout,
    int M, int N, int K) {
  constexpr int LSTR = 40;                         // 32 + 8 pad (bank-conflict free)
  __shared__ __align__(16) unsigned short As[2][128 * LSTR];
  __shared__ __align__(16) unsigned short Bs[2][128 * LSTR];

  const int tid  = threadIdx.x;
  const int lane = tid & 31;
  const int wave = tid >> 5;
  const int wm   = wave >> 2;                      // 0..1  (64 rows each)
  const int wn   = wave & 3;                       // 0..3  (32 cols each)
  const int m0   = blockIdx.y * 128;
  const int n0   = blockIdx.x * 128;
  const int nidx = lane & 15;                      // M / N index within 16x16 tile
  const int kbA  = (lane >> 4) * 8;                // A-operand K base (split +16)
  const int kbB  = (lane >> 4) * 8;                // B-operand K base (split +16)

  v8f acc[4][2] = {};

  const int srow = tid >> 1;                       // staging row 0..127
  const int scol = (tid & 1) * 16;                 // staging col 0 / 16
  const unsigned short* ga = A  + (size_t)(m0 + srow) * K + scol;
  const unsigned short* gb = Bt + (size_t)(n0 + srow) * K + scol;
  const int loff = srow * LSTR + scol;

  auto stage = [&](int buf, int k0) {
#if USE_ASYNC_LDS
    async_copy16(ga + k0,     &As[buf][loff]);
    async_copy16(ga + k0 + 8, &As[buf][loff + 8]);
    async_copy16(gb + k0,     &Bs[buf][loff]);
    async_copy16(gb + k0 + 8, &Bs[buf][loff + 8]);
#else
    const v4u a0 = *(const v4u*)(ga + k0);
    const v4u a1 = *(const v4u*)(ga + k0 + 8);
    const v4u b0 = *(const v4u*)(gb + k0);
    const v4u b1 = *(const v4u*)(gb + k0 + 8);
    *(v4u*)&As[buf][loff]     = a0;
    *(v4u*)&As[buf][loff + 8] = a1;
    *(v4u*)&Bs[buf][loff]     = b0;
    *(v4u*)&Bs[buf][loff + 8] = b1;
#endif
  };

  stage(0, 0);
  const int nk = K >> 5;
  for (int i = 0; i < nk; ++i) {
    const int cur  = i & 1;
    const bool more = (i + 1) < nk;
    if (more) stage(1 - cur, (i + 1) * 32);        // prefetch next tile
#if USE_ASYNC_LDS
    if (more) wait_async<4>();                     // current tile landed; next in flight
    else      wait_async<0>();
#endif
    __syncthreads();                               // tile visible to all waves

    const unsigned short* as = As[cur];
    const unsigned short* bs = Bs[cur];
    BFrag bf[2];
#pragma unroll
    for (int nn = 0; nn < 2; ++nn) {
      const int brow = wn * 32 + nn * 16 + nidx;
      bf[nn].u[0] = *(const v4u*)&bs[brow * LSTR + kbB];
      bf[nn].u[1] = *(const v4u*)&bs[brow * LSTR + kbB + 16];
    }
#pragma unroll
    for (int mm = 0; mm < 4; ++mm) {
      BFrag af;
      const int arow = wm * 64 + mm * 16 + nidx;
      af.u[0] = *(const v4u*)&as[arow * LSTR + kbA];
      af.u[1] = *(const v4u*)&as[arow * LSTR + kbA + 16];
      acc[mm][0] = wmma_bf16(af, bf[0], acc[mm][0]);
      acc[mm][1] = wmma_bf16(af, bf[1], acc[mm][1]);
    }
    __syncthreads();                               // done reading before overwrite
  }

  const int rowo = 8 * (lane >> 4);                // C layout: row = r + 8*(lane/16)
#pragma unroll
  for (int mm = 0; mm < 4; ++mm)
#pragma unroll
    for (int nn = 0; nn < 2; ++nn) {
      const int ncol = n0 + wn * 32 + nn * 16 + nidx;
      const float bv = bias[ncol];
#pragma unroll
      for (int r = 0; r < 8; ++r) {
        const int mrow = m0 + wm * 64 + mm * 16 + rowo + r;
        const float val = acc[mm][nn][r] + bv;
        if constexpr (MODE == 0) {
          ((float*)Cout)[(size_t)mrow * N + ncol] = val;
        } else if constexpr (MODE == 1) {
          ((unsigned short*)Cout)[(size_t)mrow * N + ncol] = f2bf(val);
        } else if constexpr (MODE == 2) {
          ((unsigned short*)Cout)[(size_t)mrow * N + ncol] = f2bf(fmaxf(val, 0.0f));
        } else {                                   // V^T: [B][H][HS][S]
          const int bb = mrow >> 10, s  = mrow & (S_ - 1);
          const int hh = ncol >> 6,  hs = ncol & (HS_ - 1);
          ((unsigned short*)Cout)[((size_t)(bb * H_ + hh) * HS_ + hs) * S_ + s] = f2bf(val);
        }
      }
    }
}

// ---------------------------------------------------------------------------
// Flash-style attention: grid (S/64, H, B), 128 threads = 4 waves,
// each wave owns 16 query rows, streams keys 32 at a time.
// Q/K: bf16 [B][S][D] (head slice), Vt: bf16 [B][H][HS][S], ctx: bf16 [B][S][D].
// ---------------------------------------------------------------------------
__global__ __launch_bounds__(128) void attn_kernel(
    const unsigned short* __restrict__ Qm, const unsigned short* __restrict__ Km,
    const unsigned short* __restrict__ Vt, const int* __restrict__ mask,
    unsigned short* __restrict__ ctx) {
  constexpr int PSTR = 40;
  __shared__ __align__(16) unsigned short ptile[4 * 16 * PSTR];

  const int lane = threadIdx.x & 31;
  const int wave = threadIdx.x >> 5;
  const int qb   = blockIdx.x * 64 + wave * 16;
  const int h    = blockIdx.y;
  const int b    = blockIdx.z;
  const int nidx = lane & 15;
  const int kbA  = (lane >> 4) * 8;
  const int kbB  = (lane >> 4) * 8;

  const size_t headOff = (size_t)b * S_ * D_ + (size_t)h * HS_;

  BFrag qa[2];                                     // Q rows, hs 0..31 / 32..63
  {
    const unsigned short* qp = Qm + headOff + (size_t)(qb + nidx) * D_;
#pragma unroll
    for (int kk = 0; kk < 2; ++kk) {
      qa[kk].u[0] = *(const v4u*)(qp + kk * 32 + kbA);
      qa[kk].u[1] = *(const v4u*)(qp + kk * 32 + kbA + 16);
    }
  }

  v8f accv[4] = {};
  float mrun[8], lrun[8];
#pragma unroll
  for (int r = 0; r < 8; ++r) { mrun[r] = -3.0e38f; lrun[r] = 0.0f; }

  unsigned short* pt = &ptile[wave * 16 * PSTR];

  for (int kt = 0; kt < S_; kt += 32) {
    // ---- scores: two 16x16 tiles (keys kt..+15, kt+16..+31) -----------------
    v8f s0 = {}, s1 = {};
    {
      const unsigned short* kp = Km + headOff + (size_t)(kt + nidx) * D_;
      BFrag kf;
#pragma unroll
      for (int kk = 0; kk < 2; ++kk) {
        kf.u[0] = *(const v4u*)(kp + kk * 32 + kbB);
        kf.u[1] = *(const v4u*)(kp + kk * 32 + kbB + 16);
        s0 = wmma_bf16(qa[kk], kf, s0);
      }
      const unsigned short* kp1 = kp + 16 * D_;
#pragma unroll
      for (int kk = 0; kk < 2; ++kk) {
        kf.u[0] = *(const v4u*)(kp1 + kk * 32 + kbB);
        kf.u[1] = *(const v4u*)(kp1 + kk * 32 + kbB + 16);
        s1 = wmma_bf16(qa[kk], kf, s1);
      }
    }
    const int mk0 = mask[b * S_ + kt + nidx];
    const int mk1 = mask[b * S_ + kt + 16 + nidx];
#pragma unroll
    for (int r = 0; r < 8; ++r) {
      s0[r] = mk0 ? s0[r] * 0.125f : -1.0e18f;     // scale = 1/sqrt(64)
      s1[r] = mk1 ? s1[r] * 0.125f : -1.0e18f;
    }

    // ---- online softmax (row reductions across 16-lane halves) -------------
    float tm[8], corr[8], p0[8], p1[8], ps[8];
#pragma unroll
    for (int r = 0; r < 8; ++r) tm[r] = fmaxf(s0[r], s1[r]);
#pragma unroll
    for (int d = 1; d < 16; d <<= 1)
#pragma unroll
      for (int r = 0; r < 8; ++r) tm[r] = fmaxf(tm[r], __shfl_xor(tm[r], d, 32));
#pragma unroll
    for (int r = 0; r < 8; ++r) {
      const float mnew = fmaxf(mrun[r], tm[r]);
      corr[r] = __expf(mrun[r] - mnew);
      mrun[r] = mnew;
      p0[r] = __expf(s0[r] - mnew);
      p1[r] = __expf(s1[r] - mnew);
      ps[r] = p0[r] + p1[r];
    }
#pragma unroll
    for (int d = 1; d < 16; d <<= 1)
#pragma unroll
      for (int r = 0; r < 8; ++r) ps[r] += __shfl_xor(ps[r], d, 32);
#pragma unroll
    for (int r = 0; r < 8; ++r) lrun[r] = lrun[r] * corr[r] + ps[r];
#pragma unroll
    for (int t = 0; t < 4; ++t)
#pragma unroll
      for (int r = 0; r < 8; ++r) accv[t][r] *= corr[r];

    // ---- redistribute P (C-layout) -> A-operand layout via LDS -------------
    __syncthreads();
    const int prow0 = 8 * (lane >> 4);
#pragma unroll
    for (int r = 0; r < 8; ++r) {
      pt[(prow0 + r) * PSTR + nidx]      = f2bf(p0[r]);
      pt[(prow0 + r) * PSTR + 16 + nidx] = f2bf(p1[r]);
    }
    __syncthreads();
    BFrag pa;
    pa.u[0] = *(const v4u*)&pt[nidx * PSTR + kbA];
    pa.u[1] = *(const v4u*)&pt[nidx * PSTR + kbA + 16];

    // ---- ctx += P @ V (V^T layout gives contiguous-K B fragments) ----------
    const unsigned short* vp = Vt + ((size_t)(b * H_ + h) * HS_) * S_ + kt;
#pragma unroll
    for (int t = 0; t < 4; ++t) {
      BFrag vf;
      const unsigned short* vpp = vp + (size_t)(t * 16 + nidx) * S_;
      vf.u[0] = *(const v4u*)(vpp + kbB);
      vf.u[1] = *(const v4u*)(vpp + kbB + 16);
      accv[t] = wmma_bf16(pa, vf, accv[t]);
    }
  }

  const int rowo = 8 * (lane >> 4);
#pragma unroll
  for (int t = 0; t < 4; ++t)
#pragma unroll
    for (int r = 0; r < 8; ++r) {
      const float val = accv[t][r] / lrun[r];
      const int row = qb + rowo + r;
      const int col = h * HS_ + t * 16 + nidx;
      ctx[((size_t)b * S_ + row) * D_ + col] = f2bf(val);
    }
}

// ---------------------------------------------------------------------------
// Fused residual-add + LayerNorm; emits f32 stream and bf16 GEMM input.
// One block (256 threads) per token row of D=1024.
// ---------------------------------------------------------------------------
__global__ __launch_bounds__(256) void ln_res_kernel(
    const float* __restrict__ resid, const float* __restrict__ proj,
    const float* __restrict__ gamma, const float* __restrict__ beta,
    float* __restrict__ xout, unsigned short* __restrict__ xbout) {
  __shared__ float red[8];
  const int t = threadIdx.x;
  const size_t base = (size_t)blockIdx.x * D_;

  float v[4];
  float s = 0.0f;
#pragma unroll
  for (int i = 0; i < 4; ++i) {
    const int c = t + i * 256;
    v[i] = resid[base + c] + proj[base + c];
    s += v[i];
  }
  for (int d = 16; d > 0; d >>= 1) s += __shfl_xor(s, d, 32);
  if ((t & 31) == 0) red[t >> 5] = s;
  __syncthreads();
  float tot = 0.0f;
#pragma unroll
  for (int i = 0; i < 8; ++i) tot += red[i];
  const float mu = tot * (1.0f / D_);
  __syncthreads();

  float sq = 0.0f;
#pragma unroll
  for (int i = 0; i < 4; ++i) { const float d = v[i] - mu; sq += d * d; }
  for (int d = 16; d > 0; d >>= 1) sq += __shfl_xor(sq, d, 32);
  if ((t & 31) == 0) red[t >> 5] = sq;
  __syncthreads();
  float tots = 0.0f;
#pragma unroll
  for (int i = 0; i < 8; ++i) tots += red[i];
  const float inv = rsqrtf(tots * (1.0f / D_) + 1e-5f);

#pragma unroll
  for (int i = 0; i < 4; ++i) {
    const int c = t + i * 256;
    const float o = (v[i] - mu) * inv * gamma[c] + beta[c];
    xout [base + c] = o;
    xbout[base + c] = f2bf(o);
  }
}

// ---------------------------------------------------------------------------
// Host orchestration (graph-capture safe: only kernel launches on `stream`).
// ---------------------------------------------------------------------------
extern "C" void kernel_launch(void* const* d_in, const int* in_sizes, int n_in,
                              void* d_out, int out_size, void* d_ws, size_t ws_size,
                              hipStream_t stream) {
  const int*   toks = (const int*)d_in[0];
  const int*   mask = (const int*)d_in[1];
  const float* emb  = (const float*)d_in[2];
  const float* Wq = (const float*)d_in[3];   const float* bq = (const float*)d_in[4];
  const float* Wk = (const float*)d_in[5];   const float* bk = (const float*)d_in[6];
  const float* Wv = (const float*)d_in[7];   const float* bv = (const float*)d_in[8];
  const float* Wo = (const float*)d_in[9];   const float* bo = (const float*)d_in[10];
  const float* ln1s = (const float*)d_in[11]; const float* ln1b = (const float*)d_in[12];
  const float* W1 = (const float*)d_in[13];  const float* b1 = (const float*)d_in[14];
  const float* W2 = (const float*)d_in[15];  const float* b2 = (const float*)d_in[16];
  const float* ln2s = (const float*)d_in[17]; const float* ln2b = (const float*)d_in[18];

  char* wp = (char*)d_ws;
  auto carve = [&](size_t bytes) {
    char* p = wp;
    wp += (bytes + 255) & ~(size_t)255;
    return p;
  };
  const size_t szDD = (size_t)D_ * D_;
  const size_t szDF = (size_t)D_ * FF_;

  unsigned short* WqT = (unsigned short*)carve(L_ * szDD * 2);
  unsigned short* WkT = (unsigned short*)carve(L_ * szDD * 2);
  unsigned short* WvT = (unsigned short*)carve(L_ * szDD * 2);
  unsigned short* WoT = (unsigned short*)carve(L_ * szDD * 2);
  unsigned short* W1T = (unsigned short*)carve(L_ * szDF * 2);
  unsigned short* W2T = (unsigned short*)carve(L_ * szDF * 2);
  float*          x   = (float*)carve((size_t)MTOT * D_ * 4);
  float*          tmp = (float*)carve((size_t)MTOT * D_ * 4);
  unsigned short* xb  = (unsigned short*)carve((size_t)MTOT * D_ * 2);
  unsigned short* qB  = (unsigned short*)carve((size_t)MTOT * D_ * 2);
  unsigned short* kB  = (unsigned short*)carve((size_t)MTOT * D_ * 2);
  unsigned short* vT  = (unsigned short*)carve((size_t)MTOT * D_ * 2);
  unsigned short* cB  = (unsigned short*)carve((size_t)MTOT * D_ * 2);
  unsigned short* fH  = (unsigned short*)carve((size_t)MTOT * FF_ * 2);

  const int cvtDD = (int)((szDD + 255) / 256);
  const int cvtDF = (int)((szDF + 255) / 256);
  for (int l = 0; l < L_; ++l) {
    wcvt_kernel<<<cvtDD, 256, 0, stream>>>(Wq + l * szDD, WqT + l * szDD, D_, D_);
    wcvt_kernel<<<cvtDD, 256, 0, stream>>>(Wk + l * szDD, WkT + l * szDD, D_, D_);
    wcvt_kernel<<<cvtDD, 256, 0, stream>>>(Wv + l * szDD, WvT + l * szDD, D_, D_);
    wcvt_kernel<<<cvtDD, 256, 0, stream>>>(Wo + l * szDD, WoT + l * szDD, D_, D_);
    wcvt_kernel<<<cvtDF, 256, 0, stream>>>(W1 + l * szDF, W1T + l * szDF, D_, FF_);
    wcvt_kernel<<<cvtDF, 256, 0, stream>>>(W2 + l * szDF, W2T + l * szDF, FF_, D_);
  }

  embed_kernel<<<MTOT, 256, 0, stream>>>(toks, emb, x, xb);

  const dim3 gDD(D_ / 128, MTOT / 128);    // N = 1024
  const dim3 gDF(FF_ / 128, MTOT / 128);   // N = 4096
  const dim3 gAttn(S_ / 64, H_, B_);

  for (int l = 0; l < L_; ++l) {
    gemm_bf16_kernel<1><<<gDD, 256, 0, stream>>>(xb, WqT + l * szDD, bq + l * D_, qB, MTOT, D_, D_);
    gemm_bf16_kernel<1><<<gDD, 256, 0, stream>>>(xb, WkT + l * szDD, bk + l * D_, kB, MTOT, D_, D_);
    gemm_bf16_kernel<3><<<gDD, 256, 0, stream>>>(xb, WvT + l * szDD, bv + l * D_, vT, MTOT, D_, D_);
    attn_kernel<<<gAttn, 128, 0, stream>>>(qB, kB, vT, mask, cB);
    gemm_bf16_kernel<0><<<gDD, 256, 0, stream>>>(cB, WoT + l * szDD, bo + l * D_, tmp, MTOT, D_, D_);
    ln_res_kernel<<<MTOT, 256, 0, stream>>>(x, tmp, ln1s + l * D_, ln1b + l * D_, x, xb);
    gemm_bf16_kernel<2><<<gDF, 256, 0, stream>>>(xb, W1T + l * szDF, b1 + l * FF_, fH, MTOT, FF_, D_);
    gemm_bf16_kernel<0><<<gDD, 256, 0, stream>>>(fH, W2T + l * szDF, b2 + l * D_, tmp, MTOT, D_, FF_);
    float* xo = (l == L_ - 1) ? (float*)d_out : x;
    ln_res_kernel<<<MTOT, 256, 0, stream>>>(x, tmp, ln2s + l * D_, ln2b + l * D_, xo, xb);
  }

  (void)in_sizes; (void)n_in; (void)out_size; (void)ws_size;
}
